// SpeakerVerificationLSTMEncoder_75342316307091
// MI455X (gfx1250) — compile-verified, matrix operand order checked by reading, not run on hardware
//
#include <hip/hip_runtime.h>
#include <hip/hip_bf16.h>

// ---------------------------------------------------------------------------
// SpeakerVerification LSTM encoder for gfx1250 (MI455X).
// 3-layer projected LSTM (proj_size=256), bf16 WMMA (v_wmma_f32_16x16x32_bf16)
// with fp32 accumulation, followed by linear + relu + L2 normalize.
//
// One persistent kernel per layer: the 160-step recurrence loops inside the
// kernel. Each block owns 16 batch rows for the whole layer: c lives in
// registers, h lives in LDS across timesteps. Weights stream from the
// L2-resident bf16 copies each step; an opaque memory clobber at the top of
// each iteration stops LICM from hoisting (and then spilling) the
// loop-invariant weight fragment loads.
// ---------------------------------------------------------------------------

#define S_  64
#define U_  10
#define T_  160
#define F_  80
#define H_  768
#define P_  256
#define E_  256
#define B_  (S_*U_)      // 640 sequences
#define G4H (4*H_)       // 3072 gate columns
#define K0PAD 96         // F=80 padded to multiple of 32

typedef __attribute__((ext_vector_type(16))) __bf16 v16bf;
typedef __attribute__((ext_vector_type(8)))  float  v8f;

__device__ __forceinline__ float sigmf(float x) { return 1.0f / (1.0f + __expf(-x)); }

union FragU { uint4 u[2]; v16bf v; };

// Global weight fragment: uniform SGPR base + ONE divergent 32-bit byte
// offset + compile-time immediate (tile stride + k offset) ->
// global_load_b128 vdst, voff, s[base] offset:imm  (GVS addressing).
__device__ __forceinline__ v16bf load_fragg(const __bf16* __restrict__ base,
                                            unsigned voff, int immBytes) {
  const char* p = (const char*)base + voff;
  FragU f;
  f.u[0] = *(const uint4*)(p + immBytes);
  f.u[1] = *(const uint4*)(p + immBytes + 32);
  return f.v;
}
// LDS fragment: base already includes lane-group element offset (grp*8);
// k0 compile-time -> immediate-offset ds_load_b128 pair.
__device__ __forceinline__ v16bf load_fragl(const __bf16* base, int k0) {
  FragU f;
  f.u[0] = *(const uint4*)(base + k0);
  f.u[1] = *(const uint4*)(base + k0 + 16);
  return f.v;
}

// ---------------------------------------------------------------------------
// Whole-layer persistent LSTM kernel.
// grid = B/16 = 40 blocks, block = 512 threads (16 waves of 32).
// Each wave owns the 12 interleaved gate N-tiles {w + 16*jj}, so tiles
// jj, jj+3, jj+6, jj+9 are the i/f/g/o gates of the SAME hidden units ->
// the nonlinear cell update is register-local and c persists in registers.
// ---------------------------------------------------------------------------
template <int KIN>   // 96 (layer0) or 256 (layers 1,2)
__global__ __launch_bounds__(512)
void lstm_layer_kernel(const __bf16* __restrict__ seq_in,   // [T][B][256] (KIN valid per row)
                       const __bf16* __restrict__ Wih,      // [3072][KIN] bf16, zero-padded
                       const __bf16* __restrict__ Whh,      // [3072][256] bf16
                       const float*  __restrict__ bias,     // [3072] = b_ih + b_hh
                       const __bf16* __restrict__ Whr,      // [256][768] bf16
                       __bf16* __restrict__ seq_out,        // [T][B][256] or nullptr
                       float*  __restrict__ h_f32)          // [B][256] final hidden
{
  __shared__ __align__(16) __bf16 actLDS[16 * 256];   // 8 KB  x_t staging
  __shared__ __align__(16) __bf16 hidLDS[16 * H_];    // 24 KB pre-projection hidden
  __shared__ __align__(16) __bf16 hLDS[16 * 256];     // 8 KB  recurrent h state

  const int tid  = threadIdx.x;
  const int lane = tid & 31;
  const int w    = tid >> 5;        // wave 0..15
  const int grp  = lane >> 4;       // lane group 0/1
  const int ln   = lane & 15;
  const int b0   = blockIdx.x * 16;

  // Single 32-bit vaddr per weight matrix; tile strides become immediates.
  constexpr int IH_TS = KIN * 32;                    // bytes per 16-col tile of Wih
  constexpr int HH_TS = 256 * 32;                    // bytes per 16-col tile of Whh
  const unsigned vIh = (unsigned)(w * IH_TS + ln * (KIN * 2) + grp * 16);
  const unsigned vHh = (unsigned)(w * HH_TS + ln * 512 + grp * 16);
  const unsigned vHr = (unsigned)((w * 16 + ln) * (H_ * 2) + grp * 16);

  const __bf16* aBase  = &actLDS[ln * KIN + grp * 8];
  const __bf16* hBase  = &hLDS[ln * 256 + grp * 8];
  const __bf16* pABase = &hidLDS[ln * H_ + grp * 8];
  const int pcol = w * 16 + ln;

  // c state lives in registers for the whole layer.
  float cReg[3][8];
#pragma unroll
  for (int jj = 0; jj < 3; ++jj)
#pragma unroll
    for (int r = 0; r < 8; ++r) cReg[jj][r] = 0.0f;

  // h0 = 0
  for (int i = tid; i < (16 * 256) / 8; i += 512)
    *(uint4*)&hLDS[i * 8] = uint4{0u, 0u, 0u, 0u};
  __syncthreads();

#pragma unroll 1
  for (int t = 0; t < T_; ++t) {
    // Opaque clobber: memory may change every iteration, so no global/LDS
    // load is loop-invariant -> LICM cannot hoist (and spill) weight loads.
    asm volatile("" ::: "memory");

    const __bf16* xin = seq_in + (size_t)t * (B_ * 256);

    // ---- stage x_t, init accumulators with bias (re-read: L2 resident) ---
    {
      constexpr int CPR = KIN >> 3;
      for (int i = tid; i < 16 * CPR; i += 512) {
        const int m = i / CPR, c = i % CPR;
        *(uint4*)&actLDS[m * KIN + c * 8] =
            *(const uint4*)&xin[(size_t)(b0 + m) * 256 + c * 8];
      }
    }
    v8f acc[12];
#pragma unroll
    for (int jj = 0; jj < 12; ++jj) {
      const float bvv = bias[(w + 16 * jj) * 16 + ln];
#pragma unroll
      for (int r = 0; r < 8; ++r) acc[jj][r] = bvv;
    }
    __syncthreads();                                    // (1) x staged

    // ---- gates += x_t @ Wih^T -------------------------------------------
#pragma unroll
    for (int k0 = 0; k0 < KIN; k0 += 32) {
      const v16bf a = load_fragl(aBase, k0);
#pragma unroll
      for (int jj = 0; jj < 12; ++jj) {
        const v16bf b = load_fragg(Wih, vIh, jj * 16 * IH_TS + k0 * 2);
        acc[jj] = __builtin_amdgcn_wmma_f32_16x16x32_bf16(
            false, a, false, b, (short)0, acc[jj], false, false);
      }
    }
    // ---- gates += h_{t-1} @ Whh^T  (A-fragments straight from hLDS) ------
#pragma unroll
    for (int k0 = 0; k0 < 256; k0 += 32) {
      const v16bf a = load_fragl(hBase, k0);
#pragma unroll
      for (int jj = 0; jj < 12; ++jj) {
        const v16bf b = load_fragg(Whh, vHh, jj * 16 * HH_TS + k0 * 2);
        acc[jj] = __builtin_amdgcn_wmma_f32_16x16x32_bf16(
            false, a, false, b, (short)0, acc[jj], false, false);
      }
    }

    // ---- cell update (register-local i/f/g/o, c in registers) ------------
#pragma unroll
    for (int jj = 0; jj < 3; ++jj) {
      const int hu = (w + 16 * jj) * 16 + ln;     // hidden unit 0..767
#pragma unroll
      for (int r = 0; r < 8; ++r) {
        const int m = r + 8 * grp;                // batch row within tile
        const float iv = sigmf(acc[jj    ][r]);
        const float fv = sigmf(acc[jj + 3][r]);
        const float gv = tanhf(acc[jj + 6][r]);
        const float ov = sigmf(acc[jj + 9][r]);
        const float cn = fv * cReg[jj][r] + iv * gv;
        cReg[jj][r] = cn;
        hidLDS[m * H_ + hu] = (__bf16)(ov * tanhf(cn));
      }
    }
    __syncthreads();                                    // (2) hidden ready, hLDS reads done
    asm volatile("" ::: "memory");                      // bound scheduling window

    // ---- projection: h_t = (o*tanh(c)) @ Whr^T  (one 16x16 tile/wave) ----
    v8f pacc;
#pragma unroll
    for (int r = 0; r < 8; ++r) pacc[r] = 0.0f;
#pragma unroll
    for (int k0 = 0; k0 < H_; k0 += 32) {
      const v16bf a = load_fragl(pABase, k0);
      const v16bf b = load_fragg(Whr, vHr, k0 * 2);
      pacc = __builtin_amdgcn_wmma_f32_16x16x32_bf16(
          false, a, false, b, (short)0, pacc, false, false);
    }

#pragma unroll
    for (int r = 0; r < 8; ++r) {
      const int m = r + 8 * grp;
      const __bf16 hv = (__bf16)pacc[r];
      hLDS[m * 256 + pcol] = hv;                        // new recurrent state
      if (seq_out)
        seq_out[(size_t)t * (B_ * 256) + (size_t)(b0 + m) * 256 + pcol] = hv;
    }
    if (t == T_ - 1) {
#pragma unroll
      for (int r = 0; r < 8; ++r) {
        const int m = r + 8 * grp;
        h_f32[(size_t)(b0 + m) * 256 + pcol] = pacc[r];
      }
    }
    __syncthreads();                                    // (3) new h visible
  }
}

// ---------------------------------------------------------------------------
// Helper kernels
// ---------------------------------------------------------------------------
__global__ void cvt_x_kernel(const float* __restrict__ x, __bf16* __restrict__ dst) {
  // x[S,U,T,F] (b = s*U+u flattened) -> dst[(t*B + b)*256 + k], k<96, zero-pad 80..95
  size_t i = (size_t)blockIdx.x * blockDim.x + threadIdx.x;
  const size_t total = (size_t)T_ * B_ * K0PAD;
  if (i >= total) return;
  const int k = (int)(i % K0PAD);
  const size_t tb = i / K0PAD;
  const int b = (int)(tb % B_);
  const int t = (int)(tb / B_);
  const float v = (k < F_) ? x[((size_t)b * T_ + t) * F_ + k] : 0.0f;
  dst[((size_t)t * B_ + b) * 256 + k] = (__bf16)v;
}

__global__ void cvt_pad_kernel(const float* __restrict__ src, __bf16* __restrict__ dst,
                               int rows, int cs, int cd) {
  size_t i = (size_t)blockIdx.x * blockDim.x + threadIdx.x;
  const size_t total = (size_t)rows * cd;
  if (i >= total) return;
  const int c = (int)(i % cd);
  const int r = (int)(i / cd);
  dst[i] = (__bf16)((c < cs) ? src[(size_t)r * cs + c] : 0.0f);
}

__global__ void bias_sum_kernel(const float* a, const float* b, float* o, int n) {
  int i = blockIdx.x * blockDim.x + threadIdx.x;
  if (i < n) o[i] = a[i] + b[i];
}

// Head: out = normalize(relu(h_last @ W_lin^T + b_lin)); one block per row.
__global__ __launch_bounds__(256)
void head_kernel(const float* __restrict__ h, const float* __restrict__ Wlin,
                 const float* __restrict__ blin, float* __restrict__ out) {
  const int row = blockIdx.x;
  const int p = threadIdx.x;
  const float* hr = h + (size_t)row * P_;
  const float* wr = Wlin + (size_t)p * P_;
  float acc = blin[p];
#pragma unroll 8
  for (int k = 0; k < P_; ++k) acc = fmaf(hr[k], wr[k], acc);
  const float v = fmaxf(acc, 0.0f);
  __shared__ float red[E_];
  red[p] = v * v;
  __syncthreads();
  for (int s = E_ / 2; s > 0; s >>= 1) {
    if (p < s) red[p] += red[p + s];
    __syncthreads();
  }
  const float norm = fmaxf(sqrtf(red[0]), 1e-12f);
  out[(size_t)row * E_ + p] = v / norm;
}

// ---------------------------------------------------------------------------
// Host launcher
// ---------------------------------------------------------------------------
extern "C" void kernel_launch(void* const* d_in, const int* in_sizes, int n_in,
                              void* d_out, int out_size, void* d_ws, size_t ws_size,
                              hipStream_t stream) {
  (void)in_sizes; (void)n_in; (void)out_size; (void)ws_size;

  const float* x = (const float*)d_in[0];
  const float* Wih_f[3] = {(const float*)d_in[1], (const float*)d_in[6],  (const float*)d_in[11]};
  const float* Whh_f[3] = {(const float*)d_in[2], (const float*)d_in[7],  (const float*)d_in[12]};
  const float* bih_f[3] = {(const float*)d_in[3], (const float*)d_in[8],  (const float*)d_in[13]};
  const float* bhh_f[3] = {(const float*)d_in[4], (const float*)d_in[9],  (const float*)d_in[14]};
  const float* Whr_f[3] = {(const float*)d_in[5], (const float*)d_in[10], (const float*)d_in[15]};
  const float* Wlin = (const float*)d_in[16];
  const float* blin = (const float*)d_in[17];
  float* out = (float*)d_out;

  // ---- carve workspace ----
  char* ws = (char*)d_ws;
  auto carve = [&](size_t bytes) -> char* {
    char* p = ws;
    ws += (bytes + 255) & ~(size_t)255;
    return p;
  };
  const size_t seq_elems = (size_t)T_ * B_ * 256;
  __bf16* buf0 = (__bf16*)carve(seq_elems * 2);
  __bf16* buf1 = (__bf16*)carve(seq_elems * 2);
  const int Kins[3] = {K0PAD, 256, 256};
  __bf16* Wih_b[3]; __bf16* Whh_b[3]; __bf16* Whr_b[3]; float* bias[3];
  for (int l = 0; l < 3; ++l) {
    Wih_b[l] = (__bf16*)carve((size_t)G4H * Kins[l] * 2);
    Whh_b[l] = (__bf16*)carve((size_t)G4H * 256 * 2);
    Whr_b[l] = (__bf16*)carve((size_t)P_ * H_ * 2);
    bias[l]  = (float*)carve((size_t)G4H * 4);
  }
  float* h_f32 = (float*)carve((size_t)B_ * P_ * 4);

  // ---- conversions (cheap, per call for determinism) ----
  {
    const size_t nx = (size_t)T_ * B_ * K0PAD;
    cvt_x_kernel<<<(int)((nx + 255) / 256), 256, 0, stream>>>(x, buf0);
  }
  for (int l = 0; l < 3; ++l) {
    const int cs = (l == 0) ? F_ : P_;
    {
      const size_t n = (size_t)G4H * Kins[l];
      cvt_pad_kernel<<<(int)((n + 255) / 256), 256, 0, stream>>>(Wih_f[l], Wih_b[l], G4H, cs, Kins[l]);
    }
    {
      const size_t n = (size_t)G4H * 256;
      cvt_pad_kernel<<<(int)((n + 255) / 256), 256, 0, stream>>>(Whh_f[l], Whh_b[l], G4H, 256, 256);
    }
    {
      const size_t n = (size_t)P_ * H_;
      cvt_pad_kernel<<<(int)((n + 255) / 256), 256, 0, stream>>>(Whr_f[l], Whr_b[l], P_, H_, H_);
    }
    bias_sum_kernel<<<(G4H + 255) / 256, 256, 0, stream>>>(bih_f[l], bhh_f[l], bias[l], G4H);
  }

  // ---- 3 persistent layer kernels (whole 160-step recurrence inside) ----
  lstm_layer_kernel<K0PAD><<<B_ / 16, 512, 0, stream>>>(
      buf0, Wih_b[0], Whh_b[0], bias[0], Whr_b[0], buf1, h_f32);
  lstm_layer_kernel<256><<<B_ / 16, 512, 0, stream>>>(
      buf1, Wih_b[1], Whh_b[1], bias[1], Whr_b[1], buf0, h_f32);
  lstm_layer_kernel<256><<<B_ / 16, 512, 0, stream>>>(
      buf0, Wih_b[2], Whh_b[2], bias[2], Whr_b[2], (__bf16*)nullptr, h_f32);

  // ---- head: linear + relu + L2 normalize ----
  head_kernel<<<B_, 256, 0, stream>>>(h_f32, Wlin, blin, out);
}
